// SAModuleBack_27547920237084
// MI455X (gfx1250) — matrix operand order repaired and checked by reading, not compiled, not run on hardware
//
#include <hip/hip_runtime.h>
#include <hip/hip_fp16.h>

#define BATCH 4
#define CH    256
#define NPIX  4096   // 64*64

typedef __attribute__((ext_vector_type(16))) _Float16 v16h;
typedef __attribute__((ext_vector_type(8)))  _Float16 v8h;
typedef __attribute__((ext_vector_type(8)))  float    v8f;
typedef __attribute__((ext_vector_type(4)))  float    v4f;

typedef union { v16h v; v8h h[2]; } V16u;   // assemble A-frag halves w/o moves

// ---------------------------------------------------------------------------
// Pass 0: f16 conversions, 8 elements/thread (v_cvt_pk_f16_f32 + b128 store)
// ---------------------------------------------------------------------------
__global__ __launch_bounds__(256)
void cvt_x_kernel(const float* __restrict__ x, _Float16* __restrict__ xh, int n8)
{
    const int i = blockIdx.x * 256 + threadIdx.x;
    if (i < n8) {
        const v4f* src = (const v4f*)x + 2 * (size_t)i;
        const v4f a = src[0], b = src[1];
        v8h o;
        o[0]=(_Float16)a.x; o[1]=(_Float16)a.y; o[2]=(_Float16)a.z; o[3]=(_Float16)a.w;
        o[4]=(_Float16)b.x; o[5]=(_Float16)b.y; o[6]=(_Float16)b.z; o[7]=(_Float16)b.w;
        *((v8h*)xh + i) = o;
    }
}

__global__ __launch_bounds__(256)
void cvt_w_kernel(const float* __restrict__ Wq, const float* __restrict__ Wk,
                  const float* __restrict__ Wv,
                  _Float16* __restrict__ whq, _Float16* __restrict__ whk,
                  _Float16* __restrict__ whv)
{
    const int i = blockIdx.x * 256 + threadIdx.x;   // CH*CH/8 = 8192 iterations
    const float* srcs[3] = {Wq, Wk, Wv};
    _Float16*    dsts[3] = {whq, whk, whv};
    #pragma unroll
    for (int p = 0; p < 3; ++p) {
        const v4f* src = (const v4f*)srcs[p] + 2 * (size_t)i;
        const v4f a = src[0], b = src[1];
        v8h o;
        o[0]=(_Float16)a.x; o[1]=(_Float16)a.y; o[2]=(_Float16)a.z; o[3]=(_Float16)a.w;
        o[4]=(_Float16)b.x; o[5]=(_Float16)b.y; o[6]=(_Float16)b.z; o[7]=(_Float16)b.w;
        *((v8h*)dsts[p] + i) = o;
    }
}

// ---------------------------------------------------------------------------
// Pass 1: WMMA projections, oriented M=out-channels, N=pixels.
//   A = f16(W) tile (table k order), B = f16(x) tile (table k order, resident),
//   D: lane holds o = obase + r + 8*hi, pixel fixed.
// q/v stored [B][N][C] (contiguous epilogue), k' = k+pos stored [B][C][N].
// A-loads are software-pipelined one K-step ahead of the WMMA chain.
// ---------------------------------------------------------------------------
__global__ __launch_bounds__(256)
void proj_wmma_kernel(const _Float16* __restrict__ xh,   // [B][C][N]
                      const _Float16* __restrict__ whq,  // [C][C] (o,c)
                      const _Float16* __restrict__ whk,
                      const _Float16* __restrict__ whv,
                      const float* __restrict__ bq, const float* __restrict__ bk,
                      const float* __restrict__ bv,
                      const float* __restrict__ rel_h, const float* __restrict__ rel_w,
                      _Float16* __restrict__ qo,  // [B][N][C]
                      _Float16* __restrict__ ko,  // [B][C][N]
                      _Float16* __restrict__ vo)  // [B][N][C]
{
    const int tiles = NPIX / 64;
    const int b  = blockIdx.x / tiles;
    const int t  = blockIdx.x % tiles;          // h coordinate of this block
    const int n0 = t * 64;
    const int wave  = threadIdx.x >> 5;
    const int lane  = threadIdx.x & 31;
    const int row16 = lane & 15;
    const int hi    = lane >> 4;
    const int klo   = hi ? 8 : 0;
    const int nt    = wave & 3;                 // pixel tile 0..3
    const int oh    = wave >> 2;                // o half 0..1
    const int npix  = n0 + nt * 16;
    const int w_c   = nt * 16 + row16;          // w coordinate of lane's pixel

    // B fragments: x rows, resident for all 24 tiles this wave computes
    v16h Bx[8];
    #pragma unroll
    for (int kb = 0; kb < 8; ++kb)
        #pragma unroll
        for (int s = 0; s < 16; ++s)
            Bx[kb][s] = xh[((size_t)b * CH + kb * 32 + klo + (s & 7) + ((s & 8) << 1)) * NPIX + npix + row16];

    #pragma unroll 1
    for (int p = 0; p < 3; ++p) {
        const _Float16* W   = (p == 0) ? whq : (p == 1) ? whk : whv;
        const float*  bias  = (p == 0) ? bq  : (p == 1) ? bk  : bv;
        #pragma unroll 1
        for (int ot8 = 0; ot8 < 8; ++ot8) {
            const int obase = oh * 128 + ot8 * 16;
            const _Float16* wr = W + (size_t)(obase + row16) * CH;
            v8f D = {};
            V16u A;
            A.h[0] = *(const v8h*)(wr + klo);
            A.h[1] = *(const v8h*)(wr + 16 + klo);
            #pragma unroll
            for (int kb = 0; kb < 8; ++kb) {
                V16u An;
                if (kb < 7) {                                    // prefetch next K-step
                    An.h[0] = *(const v8h*)(wr + (kb + 1) * 32 + klo);
                    An.h[1] = *(const v8h*)(wr + (kb + 1) * 32 + 16 + klo);
                }
                D = __builtin_amdgcn_wmma_f32_16x16x32_f16(false, A.v, false, Bx[kb],
                                                           (short)0, D, false, false);
                if (kb < 7) A = An;
            }
            const int o0 = obase + (hi << 3);   // lane's first out channel
            const int n  = npix + row16;        // lane's pixel
            if (p == 1) {
                // k' = Wk x + bk + pos -> [B][C][N] (strided over r)
                #pragma unroll
                for (int r = 0; r < 8; ++r) {
                    const int o = o0 + r;
                    const float val = D[r] + bias[o]
                                    + rel_h[o * 64 + t] + rel_w[o * 64 + w_c];
                    ko[((size_t)b * CH + o) * NPIX + n] = (_Float16)val;
                }
            } else {
                _Float16* dst = ((p == 0) ? qo : vo) + ((size_t)b * NPIX + n) * CH + o0;
                const v4f* bb = (const v4f*)(bias + o0);
                const v4f b0 = bb[0], b1 = bb[1];
                const float bl[8] = {b0.x, b0.y, b0.z, b0.w, b1.x, b1.y, b1.z, b1.w};
                #pragma unroll
                for (int r = 0; r < 8; ++r)
                    dst[r] = (_Float16)(D[r] + bl[r]);   // 8 contiguous halves
            }
        }
    }
}

// ---------------------------------------------------------------------------
// Pass 2: flash attention, wave-private, shuffle-free softmax.
// One wave = 16 query rows; K'/V stream from the 192MB L2 (24MB workspace).
// Per 32-column step:
//   16 WMMA for S=Q.K'^T (K-loads pipelined one step ahead),
//   S scattered into padded LDS; lane l and lane l+16 split row l%16's 32
//   columns exactly along the A-fragment partition, so each lane loads only
//   its 16 S values, does 16 exps, and max/sum combine with its partner
//   through two LDS slots (lane^16). P A-fragment built directly in regs.
//   16 WMMA for O += P.V (V-loads pipelined).
// O-rescale is ballot-gated (row maxima stabilize after ~ln(128) blocks).
// ---------------------------------------------------------------------------
__global__ __launch_bounds__(128)
void flash_attn_kernel(const _Float16* __restrict__ qg,
                       const _Float16* __restrict__ kg,
                       const _Float16* __restrict__ vg,
                       const float*    __restrict__ x,
                       const float*    __restrict__ gamma,
                       float*          __restrict__ out)
{
    constexpr float L2E = 1.4426950408889634f;
    const int b     = blockIdx.x >> 6;
    const int mblk  = blockIdx.x & 63;
    const int wave  = threadIdx.x >> 5;
    const int lane  = threadIdx.x & 31;
    const int m_base = mblk * 64 + wave * 16;
    const int row16  = lane & 15;
    const int hi     = lane >> 4;
    const int klo    = hi ? 8 : 0;

    __shared__ alignas(16) float Ssh[4][16 * 36];  // pad 36: conflict-free
    __shared__ alignas(16) float Par[4][64];       // [lane]=pmax, [32+lane]=psum
    __shared__ alignas(16) float Bro[4][16];       // alpha / 1/l broadcast
    float* ss  = Ssh[wave];
    float* psc = Par[wave];
    float* br  = Bro[wave];

    // Q A-fragments resident for the whole sweep
    v16h Aq[8];
    {
        const _Float16* qr = qg + ((size_t)b * NPIX + m_base + row16) * CH;
        #pragma unroll
        for (int kb = 0; kb < 8; ++kb) {
            V16u a;
            a.h[0] = *(const v8h*)(qr + kb * 32 + klo);
            a.h[1] = *(const v8h*)(qr + kb * 32 + 16 + klo);
            Aq[kb] = a.v;
        }
    }

    v8f O[16] = {};
    float m_r = -3.0e38f, l_r = 0.0f;   // stats for row row16 (dup per half)

    const _Float16* kbase = kg + (size_t)b * CH * NPIX;
    const _Float16* vbase = vg + (size_t)b * NPIX * CH;

    for (int nb = 0; nb < NPIX; nb += 32) {
        // ---- S = Q . K'^T (pipelined K-loads) ----
        v8f S0 = {}, S1 = {};
        const _Float16* krow = kbase + (size_t)klo * NPIX + nb + row16;
        v16h B0, B1;
        #pragma unroll
        for (int s = 0; s < 16; ++s) {
            B0[s] = krow[(size_t)((s & 7) + ((s & 8) << 1)) * NPIX];
            B1[s] = krow[(size_t)((s & 7) + ((s & 8) << 1)) * NPIX + 16];
        }
        #pragma unroll
        for (int kb = 0; kb < 8; ++kb) {
            v16h nB0, nB1;
            if (kb < 7) {
                const _Float16* nk = krow + (size_t)(kb + 1) * 32 * NPIX;
                #pragma unroll
                for (int s = 0; s < 16; ++s) {
                    nB0[s] = nk[(size_t)((s & 7) + ((s & 8) << 1)) * NPIX];
                    nB1[s] = nk[(size_t)((s & 7) + ((s & 8) << 1)) * NPIX + 16];
                }
            }
            S0 = __builtin_amdgcn_wmma_f32_16x16x32_f16(false, Aq[kb], false, B0,
                                                        (short)0, S0, false, false);
            S1 = __builtin_amdgcn_wmma_f32_16x16x32_f16(false, Aq[kb], false, B1,
                                                        (short)0, S1, false, false);
            if (kb < 7) { B0 = nB0; B1 = nB1; }
        }
        if (nb + 32 < NPIX) {
            __builtin_prefetch(kbase + (size_t)lane * NPIX + nb + 32, 0, 1);
            __builtin_prefetch(vbase + (size_t)(nb + 32 + lane) * CH, 0, 1);
        }

        // ---- scatter S (D layout) into per-wave LDS rows ----
        __builtin_amdgcn_wave_barrier();
        #pragma unroll
        for (int r = 0; r < 8; ++r) {
            const int rowm = r + (hi << 3);
            ss[rowm * 36 + row16]      = S0[r];
            ss[rowm * 36 + 16 + row16] = S1[r];
        }
        __builtin_amdgcn_wave_barrier();

        // ---- split-half softmax: this lane's 16 fragment columns ----
        const float* rp = ss + row16 * 36;
        const v4f c0 = *(const v4f*)(rp + klo);
        const v4f c1 = *(const v4f*)(rp + klo + 4);
        const v4f c2 = *(const v4f*)(rp + 16 + klo);
        const v4f c3 = *(const v4f*)(rp + 16 + klo + 4);
        const float sc[16] = {c0.x,c0.y,c0.z,c0.w, c1.x,c1.y,c1.z,c1.w,
                              c2.x,c2.y,c2.z,c2.w, c3.x,c3.y,c3.z,c3.w};
        float pmax = sc[0];
        #pragma unroll
        for (int j = 1; j < 16; ++j) pmax = fmaxf(pmax, sc[j]);
        psc[lane] = pmax;                         // exchange with partner lane^16
        __builtin_amdgcn_wave_barrier();
        const float mold = m_r;
        const float mnew = fmaxf(mold, fmaxf(pmax, psc[lane ^ 16]));

        float pe[16], psum = 0.0f;
        #pragma unroll
        for (int j = 0; j < 16; ++j) {
            pe[j] = exp2f((sc[j] - mnew) * L2E);
            psum += pe[j];
        }
        psc[32 + lane] = psum;
        __builtin_amdgcn_wave_barrier();
        const float rs = psum + psc[32 + (lane ^ 16)];
        const float a  = exp2f((mold - mnew) * L2E);
        l_r = l_r * a + rs;
        m_r = mnew;

        // ---- rescale O only when some row max actually moved ----
        if (__ballot(mnew > mold)) {
            __builtin_amdgcn_wave_barrier();
            if (hi == 0) br[row16] = a;
            __builtin_amdgcn_wave_barrier();
            const v4f* av = (const v4f*)(br + (hi << 3));
            const v4f a0 = av[0], a1 = av[1];
            const float al[8] = {a0.x, a0.y, a0.z, a0.w, a1.x, a1.y, a1.z, a1.w};
            #pragma unroll
            for (int tt = 0; tt < 16; ++tt)
                #pragma unroll
                for (int r = 0; r < 8; ++r) O[tt][r] *= al[r];
        }

        // ---- P A-fragment straight from this lane's exps ----
        v16h Ap;
        #pragma unroll
        for (int j = 0; j < 16; ++j) Ap[j] = (_Float16)pe[j];

        // ---- O += P . V (pipelined V-loads) ----
        const _Float16* vr = vbase + (size_t)(nb + klo) * CH + row16;
        v16h Bv;
        #pragma unroll
        for (int s = 0; s < 16; ++s) Bv[s] = vr[((s & 7) + ((s & 8) << 1)) * CH];
        #pragma unroll
        for (int ct = 0; ct < 16; ++ct) {
            v16h nBv;
            if (ct < 15)
                #pragma unroll
                for (int s = 0; s < 16; ++s) nBv[s] = vr[((s & 7) + ((s & 8) << 1)) * CH + (ct + 1) * 16];
            O[ct] = __builtin_amdgcn_wmma_f32_16x16x32_f16(false, Ap, false, Bv,
                                                           (short)0, O[ct], false, false);
            if (ct < 15) Bv = nBv;
        }
    }

    // ---- broadcast 1/l, epilogue out = gamma*O/l + x (32B runs per lane) ----
    __builtin_amdgcn_wave_barrier();
    if (hi == 0) br[row16] = 1.0f / l_r;
    __builtin_amdgcn_wave_barrier();
    const v4f* iv = (const v4f*)(br + (hi << 3));
    const v4f i0 = iv[0], i1 = iv[1];
    const float inv[8] = {i0.x, i0.y, i0.z, i0.w, i1.x, i1.y, i1.z, i1.w};

    const float g = gamma[0];
    #pragma unroll
    for (int ct = 0; ct < 16; ++ct) {
        const int c = ct * 16 + row16;
        const size_t idx = ((size_t)b * CH + c) * NPIX + m_base + (hi << 3);
        #pragma unroll
        for (int r = 0; r < 8; ++r)
            out[idx + r] = g * (O[ct][r] * inv[r]) + x[idx + r];
    }
}

// ---------------------------------------------------------------------------
extern "C" void kernel_launch(void* const* d_in, const int* in_sizes, int n_in,
                              void* d_out, int out_size, void* d_ws, size_t ws_size,
                              hipStream_t stream)
{
    (void)in_sizes; (void)n_in; (void)out_size; (void)ws_size;
    const float* x  = (const float*)d_in[0];
    const float* Wq = (const float*)d_in[1];
    const float* bq = (const float*)d_in[2];
    const float* Wk = (const float*)d_in[3];
    const float* bk = (const float*)d_in[4];
    const float* Wv = (const float*)d_in[5];
    const float* bv = (const float*)d_in[6];
    const float* rh = (const float*)d_in[7];
    const float* rw = (const float*)d_in[8];
    const float* gm = (const float*)d_in[9];
    float* out = (float*)d_out;

    const size_t elems = (size_t)BATCH * NPIX * CH;      // 4Mi
    _Float16* qw  = (_Float16*)d_ws;                     // 8 MB
    _Float16* kw  = qw + elems;                          // 8 MB
    _Float16* vw  = kw + elems;                          // 8 MB
    _Float16* xh  = vw + elems;                          // 8 MB
    _Float16* whq = xh + elems;                          // 128 KB
    _Float16* whk = whq + (size_t)CH * CH;
    _Float16* whv = whk + (size_t)CH * CH;               // total ~32.4 MB, L2-resident

    cvt_x_kernel<<<(int)(elems / 8 / 256), 256, 0, stream>>>(x, xh, (int)(elems / 8));
    cvt_w_kernel<<<(CH * CH / 8) / 256, 256, 0, stream>>>(Wq, Wk, Wv, whq, whk, whv);
    proj_wmma_kernel<<<BATCH * (NPIX / 64), 256, 0, stream>>>(
        xh, whq, whk, whv, bq, bk, bv, rh, rw, qw, kw, vw);
    flash_attn_kernel<<<BATCH * (NPIX / 64), 128, 0, stream>>>(
        qw, kw, vw, x, gm, out);
}
